// EBT_GRC_noslice_41394894799300
// MI455X (gfx1250) — matrix operand order repaired and checked by reading, not compiled, not run on hardware
//
#include <hip/hip_runtime.h>
#include <hip/hip_bf16.h>
#include <math.h>

typedef __attribute__((ext_vector_type(16))) _Float16 v16h;
typedef __attribute__((ext_vector_type(8)))  _Float16 v8h;
typedef __attribute__((ext_vector_type(8)))  float    v8f;

#define NBATCH 256
#define SEQ    32
#define DIM    512
#define CHID   2048
#define BEAM   5

__device__ __forceinline__ float gelu_exact(float v) {
    return 0.5f * v * (1.0f + erff(v * 0.70710678118654752f));
}

// ---------------------------------------------------------------------------
// Fragment loaders (wave32) VGPR layouts, this code's contiguous k order (same in A and B):
//   A  lane l (m = l&15, sub = l>>4): half j -> K = k0+16*sub+j
//   B  lane l (n = l&15, sub = l>>4): half j -> K = k0+16*sub+j
//   C  vgpr r: m = r + 8*sub, n = l&15
// A is fp32 (converted to f16 on load); B is pre-transposed f16 (Nc x K).
// ---------------------------------------------------------------------------
__device__ __forceinline__ v16h load_a_frag(const float* __restrict__ Arow,
                                            int k0, int sub)
{
    const float* pa0 = Arow + k0 + 16 * sub;
    const float* pa1 = Arow + k0 + 16 * sub + 8;
    float4 fa0 = *(const float4*)(pa0);
    float4 fa1 = *(const float4*)(pa0 + 4);
    float4 fb0 = *(const float4*)(pa1);
    float4 fb1 = *(const float4*)(pa1 + 4);
    v16h a;
    a[0]  = (_Float16)fa0.x; a[1]  = (_Float16)fa0.y;
    a[2]  = (_Float16)fa0.z; a[3]  = (_Float16)fa0.w;
    a[4]  = (_Float16)fa1.x; a[5]  = (_Float16)fa1.y;
    a[6]  = (_Float16)fa1.z; a[7]  = (_Float16)fa1.w;
    a[8]  = (_Float16)fb0.x; a[9]  = (_Float16)fb0.y;
    a[10] = (_Float16)fb0.z; a[11] = (_Float16)fb0.w;
    a[12] = (_Float16)fb1.x; a[13] = (_Float16)fb1.y;
    a[14] = (_Float16)fb1.z; a[15] = (_Float16)fb1.w;
    return a;
}

__device__ __forceinline__ v16h load_b_frag(const _Float16* __restrict__ Bcol,
                                            int k0)
{
    v8h b0 = *(const v8h*)(Bcol + k0);
    v8h b1 = *(const v8h*)(Bcol + k0 + 8);
    v16h b;
#pragma unroll
    for (int j = 0; j < 8; ++j) { b[j] = b0[j]; b[8 + j] = b1[j]; }
    return b;
}

__device__ __forceinline__ void load_frags(const float* __restrict__ Arow,
                                           const _Float16* const bcol[4],
                                           int k0, int sub,
                                           v16h& a, v16h b[4])
{
    a = load_a_frag(Arow, k0, sub);
#pragma unroll
    for (int t = 0; t < 4; ++t) b[t] = load_b_frag(bcol[t], k0);
}

__device__ __forceinline__ void wmma_quad(const v16h& a, const v16h b[4], v8f acc[4])
{
#pragma unroll
    for (int t = 0; t < 4; ++t)
        acc[t] = __builtin_amdgcn_wmma_f32_16x16x32_f16(
            false, a, false, b[t], (short)0, acc[t], false, false);
}

// ---------------------------------------------------------------------------
// Wave-level 16(M) x 64(N) WMMA tile.  K must be a multiple of 64 (all call
// sites use K in {512, 1024, 2048}).  Software pipelined with ping-pong
// fragment sets: even k-chunks compute from set 0 while set 1 prefetches and
// vice-versa -> loads for the next chunk are always in flight during the
// WMMAs and there is no register rotation on the loop backedge.
// ---------------------------------------------------------------------------
__device__ __forceinline__ void wmma_tile_accum(const float* __restrict__ Arow,
                                                const _Float16* __restrict__ BT,
                                                int K, int nbase, int lane,
                                                v8f acc[4])
{
    const int sub = lane >> 4;
    const int l15 = lane & 15;
    const _Float16* bcol[4];
#pragma unroll
    for (int t = 0; t < 4; ++t)
        bcol[t] = BT + (size_t)(nbase + 16 * t + l15) * K + 16 * sub;

    v16h a0, a1;
    v16h b0[4], b1[4];
    load_frags(Arow, bcol, 0, sub, a0, b0);          // set 0 <- k = 0

    for (int k0 = 0; k0 < K; k0 += 64) {
        load_frags(Arow, bcol, k0 + 32, sub, a1, b1);   // set 1 <- k+32
        wmma_quad(a0, b0, acc);                         // compute k
        if (k0 + 64 < K)
            load_frags(Arow, bcol, k0 + 64, sub, a0, b0); // set 0 <- k+64
        wmma_quad(a1, b1, acc);                         // compute k+32
    }
}

// ---------------------------------------------------------------------------
// Generic GEMM: C[M x Nc] = act(A[M x K] @ W + bias).  GELU=1 -> exact gelu.
// Block: 128 threads = 4 waves; wave w handles row-tile blockIdx.y*4+w,
// 64 output columns at blockIdx.x*64.  __launch_bounds__(...,1) lifts the
// register cap so both ping-pong fragment sets stay resident (no scratch
// spills in the inner loop).
// ---------------------------------------------------------------------------
template <int GELU>
__global__ void __launch_bounds__(128, 1)
gemm_bias_kernel(const float* __restrict__ A,
                 const _Float16* __restrict__ BT,
                 const float* __restrict__ bias,
                 float* __restrict__ C,
                 int M, int K, int Nc)
{
    const int lane  = threadIdx.x & 31;
    const int wave  = threadIdx.x >> 5;
    const int mtile = blockIdx.y * 4 + wave;
    if (mtile * 16 >= M) return;
    const int nbase = blockIdx.x * 64;
    const int l15 = lane & 15, sub = lane >> 4;
    const int row = mtile * 16 + l15;
    const float* Arow = A + (size_t)row * K;

    const v8f vzero = {0.f, 0.f, 0.f, 0.f, 0.f, 0.f, 0.f, 0.f};
    v8f acc[4];
#pragma unroll
    for (int t = 0; t < 4; ++t) acc[t] = vzero;

    wmma_tile_accum(Arow, BT, K, nbase, lane, acc);

#pragma unroll
    for (int t = 0; t < 4; ++t) {
        const int n = nbase + 16 * t + l15;
        const float bv = bias[n];
#pragma unroll
        for (int r = 0; r < 8; ++r) {
            const int m = mtile * 16 + r + 8 * sub;
            float v = acc[t][r] + bv;
            if (GELU) v = gelu_exact(v);
            if (m < M) C[(size_t)m * Nc + n] = v;
        }
    }
}

// ---------------------------------------------------------------------------
// Fused pair-score kernel: for row (n,b,s), A = [content[n,b,s,:], content[n,b,s+1,:]]
// (contiguous 1024 floats).  w = gelu(A@Wd1 + bd1) . Wd2 + bd2.
// Block = 256 threads (8 waves, full 512 hidden cols), 16 rows per block.
// Deterministic fixed-order LDS reduction (no atomics).
// ---------------------------------------------------------------------------
__global__ void __launch_bounds__(256, 1)
score_kernel(const float* __restrict__ content,
             const _Float16* __restrict__ Wd1T,
             const float* __restrict__ bd1,
             const float* __restrict__ Wd2,
             const float* __restrict__ bd2,
             float* __restrict__ wout,
             int B, int Sp, int L)
{
    __shared__ float part[16][128];
    const int lane = threadIdx.x & 31;
    const int wave = threadIdx.x >> 5;   // 0..7
    const int l15 = lane & 15, sub = lane >> 4;
    const int mtile = blockIdx.x;
    const int row = mtile * 16 + l15;    // M = 256*B*Sp, always multiple of 16
    const int n  = row / (B * Sp);
    const int rm = row % (B * Sp);
    const int b  = rm / Sp;
    const int s  = rm % Sp;
    const float* Arow = content + (((size_t)(n * B + b)) * L + s) * DIM;

    const v8f vzero = {0.f, 0.f, 0.f, 0.f, 0.f, 0.f, 0.f, 0.f};
    v8f acc[4];
#pragma unroll
    for (int t = 0; t < 4; ++t) acc[t] = vzero;

    wmma_tile_accum(Arow, Wd1T, 2 * DIM, wave * 64, lane, acc);

#pragma unroll
    for (int r = 0; r < 8; ++r) {
        float p = 0.f;
#pragma unroll
        for (int t = 0; t < 4; ++t) {
            const int nn = wave * 64 + 16 * t + l15;
            p += gelu_exact(acc[t][r] + bd1[nn]) * Wd2[nn];
        }
        part[r + 8 * sub][wave * 16 + l15] = p;
    }
    __syncthreads();
    if (threadIdx.x < 16) {
        float ssum = 0.f;
        for (int q = 0; q < 128; ++q) ssum += part[threadIdx.x][q];
        wout[mtile * 16 + threadIdx.x] = ssum + bd2[0];
    }
}

// ---------------------------------------------------------------------------
// LayerNorm rows of DIM=512: Y = (X-mu)/sqrt(var+1e-5)*g + b.  One block/row.
// ---------------------------------------------------------------------------
__global__ void ln_kernel(const float* __restrict__ X, const float* __restrict__ g,
                          const float* __restrict__ bb, float* __restrict__ Y)
{
    __shared__ float red[256], red2[256];
    const int row = blockIdx.x;
    const int tid = threadIdx.x;
    const float* x = X + (size_t)row * DIM;
    float v0 = x[tid], v1 = x[tid + 256];
    red[tid] = v0 + v1;
    red2[tid] = v0 * v0 + v1 * v1;
    __syncthreads();
    for (int off = 128; off > 0; off >>= 1) {
        if (tid < off) { red[tid] += red[tid + off]; red2[tid] += red2[tid + off]; }
        __syncthreads();
    }
    const float mu = red[0] * (1.0f / DIM);
    const float var = red2[0] * (1.0f / DIM) - mu * mu;
    const float inv = rsqrtf(var + 1e-5f);
    float* y = Y + (size_t)row * DIM;
    y[tid]       = (v0 - mu) * inv * g[tid] + bb[tid];
    y[tid + 256] = (v1 - mu) * inv * g[tid + 256] + bb[tid + 256];
}

// ---------------------------------------------------------------------------
// Per-sample masked top-k + beam expansion + beam pruning (exact reference
// semantics incl. tie->lowest-index, done gating, bsel gather). 1 thread / n.
// ---------------------------------------------------------------------------
__global__ void topk_kernel(const float* __restrict__ wbuf,
                            const float* __restrict__ mask,
                            float* __restrict__ accu, float* __restrict__ bmask,
                            int* __restrict__ parent, int* __restrict__ jsel,
                            int B, int Sp, int iter, int topk)
{
    const int n = blockIdx.x * blockDim.x + threadIdx.x;
    if (n >= NBATCH) return;
    const float NEG = -1e9f;
    const float done = 1.0f - mask[n * SEQ + iter + 1];

    float ca[BEAM * BEAM], cbm[BEAM * BEAM];
    int cp[BEAM * BEAM], cj[BEAM * BEAM];
    const int nc = B * topk;

    for (int b = 0; b < B; ++b) {
        const float* wrow = wbuf + ((size_t)n * B + b) * Sp;
        float ml[SEQ], e[SEQ];
        float mx = -INFINITY;
        for (int s2 = 0; s2 < Sp; ++s2) {
            const float mk = mask[n * SEQ + iter + 1 + s2];
            ml[s2] = (mk > 0.f) ? wrow[s2] : NEG;
            mx = fmaxf(mx, ml[s2]);
        }
        float sum = 0.f;
        for (int s2 = 0; s2 < Sp; ++s2) {
            const float mk = mask[n * SEQ + iter + 1 + s2];
            e[s2] = expf(ml[s2] - mx) * mk;
            sum += e[s2];
        }
        bool used[SEQ];
        for (int s2 = 0; s2 < Sp; ++s2) used[s2] = false;
        const float a0  = accu[n * BEAM + b];
        const float bm0 = bmask[n * BEAM + b];
        for (int t = 0; t < topk; ++t) {
            int idx = 0; float best = -INFINITY;
            for (int s2 = 0; s2 < Sp; ++s2)
                if (!used[s2] && ml[s2] > best) { best = ml[s2]; idx = s2; }
            used[idx] = true;
            const float soft = e[idx] / (sum + 1e-20f);
            const float ns = logf(soft + 1e-20f);
            const int ci = b * topk + t;
            ca[ci] = a0 + ns; cp[ci] = b; cj[ci] = idx;
            const float nbm = done * ((t == 0) ? 1.f : 0.f) + (1.f - done);
            cbm[ci] = bm0 * nbm;
        }
    }

    if (nc > BEAM) {
        bool used2[BEAM * BEAM];
        for (int x = 0; x < nc; ++x) used2[x] = false;
        for (int k = 0; k < BEAM; ++k) {
            int idx = 0; float best = -INFINITY;
            for (int x = 0; x < nc; ++x) {
                const float v = (cbm[x] > 0.f) ? ca[x] : NEG;
                if (!used2[x] && v > best) { best = v; idx = x; }
            }
            used2[idx] = true;
            accu[n * BEAM + k]   = ca[idx];
            bmask[n * BEAM + k]  = cbm[idx];
            parent[n * BEAM + k] = cp[idx];
            jsel[n * BEAM + k]   = cj[idx];
        }
    } else {
        for (int k = 0; k < nc; ++k) {
            accu[n * BEAM + k]   = ca[k];
            bmask[n * BEAM + k]  = cbm[k];
            parent[n * BEAM + k] = cp[k];
            jsel[n * BEAM + k]   = cj[k];
        }
    }
}

// lr[row] = [content[n,parent,j,:], content[n,parent,j+1,:]] (contiguous copy)
__global__ void gather_lr_kernel(const float* __restrict__ content,
                                 const int* __restrict__ parent,
                                 const int* __restrict__ jsel,
                                 float* __restrict__ lr, int Bprev, int L)
{
    const int rowk = blockIdx.x;          // n*BEAM + k
    const int n = rowk / BEAM;
    const int p = parent[rowk];
    const int j = jsel[rowk];
    const float* src = content + (((size_t)(n * Bprev + p)) * L + j) * DIM;
    float* dst = lr + (size_t)rowk * (2 * DIM);
    for (int d = threadIdx.x; d < 2 * DIM; d += blockDim.x) dst[d] = src[d];
}

// gate softmax over 3 channels + combine + layernorm -> newvec
__global__ void grc_finish_kernel(const float* __restrict__ cbuf,
                                  const float* __restrict__ lr,
                                  const float* __restrict__ g2v,
                                  const float* __restrict__ b2v,
                                  float* __restrict__ newvec)
{
    __shared__ float red[256], red2[256];
    const int rowk = blockIdx.x;
    const int tid = threadIdx.x;
    const float* c = cbuf + (size_t)rowk * (4 * DIM);
    const float* l = lr + (size_t)rowk * (2 * DIM);
    const float* r = l + DIM;
    float o[2];
#pragma unroll
    for (int i = 0; i < 2; ++i) {
        const int d = tid + i * 256;
        const float a0 = c[d], a1 = c[DIM + d], a2 = c[2 * DIM + d], a3 = c[3 * DIM + d];
        const float mx = fmaxf(a0, fmaxf(a1, a2));
        const float e0 = expf(a0 - mx), e1 = expf(a1 - mx), e2 = expf(a2 - mx);
        o[i] = (e0 * l[d] + e1 * r[d] + e2 * a3) / (e0 + e1 + e2);
    }
    red[tid] = o[0] + o[1];
    red2[tid] = o[0] * o[0] + o[1] * o[1];
    __syncthreads();
    for (int off = 128; off > 0; off >>= 1) {
        if (tid < off) { red[tid] += red[tid + off]; red2[tid] += red2[tid + off]; }
        __syncthreads();
    }
    const float mu = red[0] * (1.0f / DIM);
    const float var = red2[0] * (1.0f / DIM) - mu * mu;
    const float inv = rsqrtf(var + 1e-5f);
    float* y = newvec + (size_t)rowk * DIM;
    y[tid]       = (o[0] - mu) * inv * g2v[tid] + b2v[tid];
    y[tid + 256] = (o[1] - mu) * inv * g2v[tid + 256] + b2v[tid + 256];
}

// Rebuild content: remove pair (j,j+1) of parent beam, insert newvec at j,
// blended with done = mask[n, iter+1] as in the reference.
__global__ void compose_kernel(const float* __restrict__ Cprev,
                               float* __restrict__ Cnext,
                               const int* __restrict__ parent,
                               const int* __restrict__ jsel,
                               const float* __restrict__ newvec,
                               const float* __restrict__ mask,
                               int iter, int Bprev, int L, int Lnew)
{
    const int s = blockIdx.x;
    const int rowk = blockIdx.y;          // n*BEAM + k
    const int n = rowk / BEAM;
    const int p = parent[rowk];
    const int j = jsel[rowk];
    const float done = mask[n * SEQ + iter + 1];
    const float* base = Cprev + ((size_t)(n * Bprev + p)) * L * DIM;
    const float* oldv = base + (size_t)s * DIM;
    const float* compv = (s < j) ? oldv
                       : (s == j ? (newvec + (size_t)rowk * DIM)
                                 : base + (size_t)(s + 1) * DIM);
    float* dst = Cnext + (((size_t)rowk) * Lnew + s) * DIM;
    for (int d = threadIdx.x; d < DIM; d += blockDim.x)
        dst[d] = done * compv[d] + (1.f - done) * oldv[d];
}

__global__ void final_pairs_kernel(int* __restrict__ parent, int* __restrict__ jsel)
{
    const int rk = blockIdx.x * blockDim.x + threadIdx.x;
    if (rk < NBATCH * BEAM) { parent[rk] = rk % BEAM; jsel[rk] = 0; }
}

__global__ void init_state_kernel(float* __restrict__ accu, float* __restrict__ bmask)
{
    const int i = blockIdx.x * blockDim.x + threadIdx.x;
    if (i < NBATCH * BEAM) { accu[i] = 0.f; bmask[i] = 1.f; }
}

__global__ void final_reduce_kernel(const float* __restrict__ Cfin,
                                    const float* __restrict__ accu,
                                    const float* __restrict__ bmask,
                                    float* __restrict__ out)
{
    const int n = blockIdx.x;
    float sc[BEAM], wk[BEAM];
    float mx = -INFINITY;
    for (int k = 0; k < BEAM; ++k) {
        const float bm = bmask[n * BEAM + k];
        sc[k] = bm * accu[n * BEAM + k] + (1.f - bm) * (-999999.0f);
        mx = fmaxf(mx, sc[k]);
    }
    float den = 0.f;
    for (int k = 0; k < BEAM; ++k) { wk[k] = expf(sc[k] - mx); den += wk[k]; }
    const float rden = 1.0f / den;
    for (int d = threadIdx.x; d < DIM; d += blockDim.x) {
        float a = 0.f;
        for (int k = 0; k < BEAM; ++k)
            a += wk[k] * Cfin[((size_t)(n * BEAM + k)) * DIM + d];
        out[(size_t)n * DIM + d] = a * rden;
    }
}

// W (K x Nc) fp32 -> WT (Nc x K) f16
__global__ void transpose_f16_kernel(const float* __restrict__ W,
                                     _Float16* __restrict__ WT, int K, int Nc)
{
    const int idx = blockIdx.x * blockDim.x + threadIdx.x;
    if (idx >= K * Nc) return;
    const int k = idx / Nc, n2 = idx % Nc;
    WT[(size_t)n2 * K + k] = (_Float16)W[idx];
}

// ---------------------------------------------------------------------------
extern "C" void kernel_launch(void* const* d_in, const int* in_sizes, int n_in,
                              void* d_out, int out_size, void* d_ws, size_t ws_size,
                              hipStream_t stream)
{
    (void)in_sizes; (void)n_in; (void)out_size; (void)ws_size;
    const float* x    = (const float*)d_in[0];
    const float* mask = (const float*)d_in[1];
    const float* Wi   = (const float*)d_in[2];
    const float* bi   = (const float*)d_in[3];
    const float* g1   = (const float*)d_in[4];
    const float* b1   = (const float*)d_in[5];
    const float* Wd1  = (const float*)d_in[6];
    const float* bd1  = (const float*)d_in[7];
    const float* Wd2  = (const float*)d_in[8];
    const float* bd2  = (const float*)d_in[9];
    const float* Wc1  = (const float*)d_in[10];
    const float* bc1  = (const float*)d_in[11];
    const float* Wc2  = (const float*)d_in[12];
    const float* bc2  = (const float*)d_in[13];
    const float* g2   = (const float*)d_in[14];
    const float* b2   = (const float*)d_in[15];
    float* out = (float*)d_out;

    char* ws = (char*)d_ws;
    size_t off = 0;
    auto alloc = [&](size_t bytes) -> char* {
        char* p = ws + off;
        off = (off + bytes + 255) & ~(size_t)255;
        return p;
    };
    _Float16* WiT  = (_Float16*)alloc((size_t)512 * 512 * 2);
    _Float16* Wd1T = (_Float16*)alloc((size_t)512 * 1024 * 2);
    _Float16* Wc1T = (_Float16*)alloc((size_t)2048 * 1024 * 2);
    _Float16* Wc2T = (_Float16*)alloc((size_t)2048 * 2048 * 2);
    float* tmp0  = (float*)alloc((size_t)NBATCH * SEQ * DIM * 4);
    float* contA = (float*)alloc((size_t)NBATCH * BEAM * SEQ * DIM * 4);
    float* contB = (float*)alloc((size_t)NBATCH * BEAM * SEQ * DIM * 4);
    float* wbuf  = (float*)alloc((size_t)NBATCH * BEAM * (SEQ - 1) * 4);
    float* lr    = (float*)alloc((size_t)NBATCH * BEAM * 2 * DIM * 4);
    float* hbuf  = (float*)alloc((size_t)NBATCH * BEAM * CHID * 4);
    float* cbuf  = (float*)alloc((size_t)NBATCH * BEAM * 4 * DIM * 4);
    float* newv  = (float*)alloc((size_t)NBATCH * BEAM * DIM * 4);
    float* accu  = (float*)alloc((size_t)NBATCH * BEAM * 4);
    float* bmk   = (float*)alloc((size_t)NBATCH * BEAM * 4);
    int* parent  = (int*)alloc((size_t)NBATCH * BEAM * 4);
    int* jsel    = (int*)alloc((size_t)NBATCH * BEAM * 4);

    // ---- weight prep (f16, transposed to N x K) ----
    transpose_f16_kernel<<<(512 * 512 + 255) / 256, 256, 0, stream>>>(Wi, WiT, 512, 512);
    transpose_f16_kernel<<<(1024 * 512 + 255) / 256, 256, 0, stream>>>(Wd1, Wd1T, 1024, 512);
    transpose_f16_kernel<<<(1024 * 2048 + 255) / 256, 256, 0, stream>>>(Wc1, Wc1T, 1024, 2048);
    transpose_f16_kernel<<<(2048 * 2048 + 255) / 256, 256, 0, stream>>>(Wc2, Wc2T, 2048, 2048);
    init_state_kernel<<<(NBATCH * BEAM + 255) / 256, 256, 0, stream>>>(accu, bmk);

    // ---- initial projection + layernorm: content0 = LN(x@Wi+bi)*g1+b1 ----
    {
        const int M = NBATCH * SEQ;   // 8192
        gemm_bias_kernel<0><<<dim3(DIM / 64, M / 64), 128, 0, stream>>>(
            x, WiT, bi, tmp0, M, DIM, DIM);
        ln_kernel<<<M, 256, 0, stream>>>(tmp0, g1, b1, contA);
    }

    float* cur = contA;
    float* nxt = contB;
    int B = 1, L = SEQ;
    const int Mgrc = NBATCH * BEAM;   // 1280

    for (int i = 0; i < SEQ - 2; ++i) {
        const int Sp = L - 1;
        const int tk = (Sp < BEAM) ? Sp : BEAM;
        // pair scores w[n,b,s]
        score_kernel<<<16 * B * Sp, 256, 0, stream>>>(
            cur, Wd1T, bd1, Wd2, bd2, wbuf, B, Sp, L);
        // top-k + beam expansion/pruning -> parent/jsel/accu/bmask
        topk_kernel<<<4, 64, 0, stream>>>(wbuf, mask, accu, bmk, parent, jsel,
                                          B, Sp, i, tk);
        // grc on selected pairs
        gather_lr_kernel<<<Mgrc, 128, 0, stream>>>(cur, parent, jsel, lr, B, L);
        gemm_bias_kernel<1><<<dim3(CHID / 64, Mgrc / 64), 128, 0, stream>>>(
            lr, Wc1T, bc1, hbuf, Mgrc, 2 * DIM, CHID);
        gemm_bias_kernel<0><<<dim3((4 * DIM) / 64, Mgrc / 64), 128, 0, stream>>>(
            hbuf, Wc2T, bc2, cbuf, Mgrc, CHID, 4 * DIM);
        grc_finish_kernel<<<Mgrc, 256, 0, stream>>>(cbuf, lr, g2, b2, newv);
        // rebuild sequences one shorter
        const int Lnew = L - 1;
        compose_kernel<<<dim3(Lnew, Mgrc), 256, 0, stream>>>(
            cur, nxt, parent, jsel, newv, mask, i, B, L, Lnew);
        float* t = cur; cur = nxt; nxt = t;
        L = Lnew; B = BEAM;
    }

    // ---- final merge (i = SEQ-2, L == 2): grc(content[:,:,0], content[:,:,1]) ----
    final_pairs_kernel<<<(NBATCH * BEAM + 127) / 128, 128, 0, stream>>>(parent, jsel);
    gather_lr_kernel<<<Mgrc, 128, 0, stream>>>(cur, parent, jsel, lr, BEAM, 2);
    gemm_bias_kernel<1><<<dim3(CHID / 64, Mgrc / 64), 128, 0, stream>>>(
        lr, Wc1T, bc1, hbuf, Mgrc, 2 * DIM, CHID);
    gemm_bias_kernel<0><<<dim3((4 * DIM) / 64, Mgrc / 64), 128, 0, stream>>>(
        hbuf, Wc2T, bc2, cbuf, Mgrc, CHID, 4 * DIM);
    grc_finish_kernel<<<Mgrc, 256, 0, stream>>>(cbuf, lr, g2, b2, newv);
    compose_kernel<<<dim3(1, Mgrc), 256, 0, stream>>>(
        cur, nxt, parent, jsel, newv, mask, SEQ - 2, BEAM, 2, 1);

    // ---- beam-weighted output ----
    final_reduce_kernel<<<NBATCH, 256, 0, stream>>>(nxt, accu, bmk, out);
}
